// mod_conv2d_1228360647005
// MI455X (gfx1250) — matrix operand order repaired and checked by reading, not compiled, not run on hardware
//
#include <hip/hip_runtime.h>

typedef unsigned short     u16;
typedef unsigned int       u32;
typedef unsigned long long u64;

typedef __attribute__((ext_vector_type(16))) __bf16 v16bf;
typedef __attribute__((ext_vector_type(8)))  __bf16 v8bf;
typedef __attribute__((ext_vector_type(8)))  float  v8f;

#define N_BATCH 16
#define CH      512
#define HW      64
#define PIX     (HW*HW)

__device__ __forceinline__ u16 f2bf(float f) {
    union { float f; u32 u; } v; v.f = f;
    u32 u = v.u;
    u32 r = u + 0x7FFFu + ((u >> 16) & 1u);   // round-to-nearest-even
    return (u16)(r >> 16);
}

// CDNA5 async copy: global -> LDS, 16B per lane, tracked on ASYNCcnt.
__device__ __forceinline__ void async_ld_b128(u32 lds_byte_off, const u16* gaddr) {
    asm volatile("global_load_async_to_lds_b128 %0, %1, off"
                 :: "v"(lds_byte_off), "v"((u64)(size_t)gaddr)
                 : "memory");
}
__device__ __forceinline__ void wait_async0() {
    asm volatile("s_wait_asynccnt 0" ::: "memory");
}

// ---------------- K0: BN stats -> fold into a,b:  xn = x*a[c]+b[c] ----------------
__global__ __launch_bounds__(256) void bn_stats_kernel(
    const float* __restrict__ x, const float* __restrict__ gamma,
    const float* __restrict__ beta, float* __restrict__ a, float* __restrict__ b)
{
    const int c = blockIdx.x, tid = threadIdx.x;
    float s = 0.f, s2 = 0.f;
    for (int n = 0; n < N_BATCH; ++n) {
        const float* p = x + ((size_t)n * CH + c) * PIX;
        for (int j = tid; j < PIX; j += 256) { float v = p[j]; s += v; s2 += v * v; }
    }
    __shared__ float sh[512];
    sh[tid] = s; sh[tid + 256] = s2;
    __syncthreads();
    for (int st = 128; st > 0; st >>= 1) {
        if (tid < st) { sh[tid] += sh[tid + st]; sh[256 + tid] += sh[256 + tid + st]; }
        __syncthreads();
    }
    if (tid == 0) {
        const float inv = 1.f / (float)(N_BATCH * PIX);
        float mean = sh[0] * inv;
        float var  = sh[256] * inv - mean * mean;
        float ga   = gamma[c] * rsqrtf(var + 1e-5f);
        a[c] = ga;
        b[c] = beta[c] - mean * ga;
    }
}

// ---------------- K1: scale[n,i] = y[n,:] . mod_w[i,:] + mod_b[i] ----------------
__global__ __launch_bounds__(256) void mod_scale_kernel(
    const float* __restrict__ y, const float* __restrict__ mod_w,
    const float* __restrict__ mod_b, float* __restrict__ scale)
{
    const int idx = blockIdx.x * 256 + threadIdx.x;   // 8192 = 16*512
    const int n = idx >> 9, i = idx & 511;
    const float* yr = y + n * 512;
    const float* wr = mod_w + (size_t)i * 512;
    float s = mod_b[i];
    for (int f = 0; f < 512; ++f) s = fmaf(yr[f], wr[f], s);
    scale[idx] = s;
}

// ---------------- K2a: wsq[o,i] = sum_tap weight^2 ----------------
__global__ __launch_bounds__(256) void wsq_kernel(
    const float* __restrict__ w, float* __restrict__ wsq)
{
    const int idx = blockIdx.x * 256 + threadIdx.x;   // 262144 = 512*512
    const float* p = w + (size_t)idx * 9;
    float s = 0.f;
    #pragma unroll
    for (int t = 0; t < 9; ++t) { float v = p[t]; s = fmaf(v, v, s); }
    wsq[idx] = s;
}

// ---------------- K2b: demod[n,o] = rsqrt(sum_i scale^2 * wsq + eps) ----------------
__global__ __launch_bounds__(256) void demod_kernel(
    const float* __restrict__ scale, const float* __restrict__ wsq,
    float* __restrict__ demod)
{
    const int idx = blockIdx.x * 256 + threadIdx.x;   // 8192 = 16*512
    const int n = idx >> 9, o = idx & 511;
    const float* sc = scale + n * 512;
    const float* wq = wsq + (size_t)o * 512;
    float s = 0.f;
    for (int i = 0; i < 512; ++i) { float t = sc[i]; s = fmaf(t * t, wq[i], s); }
    demod[idx] = rsqrtf(s + 1e-8f);
}

// ---------------- K3: pack weights bf16 as wpk[tap][o][i] ----------------
__global__ __launch_bounds__(256) void pack_w_kernel(
    const float* __restrict__ w, u16* __restrict__ wpk)
{
    const int idx = blockIdx.x * 256 + threadIdx.x;   // 2359296 = 9*512*512
    const int i = idx & 511, rem = idx >> 9;
    const int o = rem & 511, t = rem >> 9;
    wpk[idx] = f2bf(w[((size_t)(o * 512 + i)) * 9 + t]);
}

// ---------------- K4: xs2[n,y,x,c] = bf16( relu(x*a+b) * scale[n,c] ), channel-last ----------------
// One block per (n, y) row; LDS tile transpose for coalesced in/out.
__global__ __launch_bounds__(256) void make_xs_kernel(
    const float* __restrict__ x, const float* __restrict__ a,
    const float* __restrict__ b, const float* __restrict__ scale,
    u16* __restrict__ xs2)
{
    __shared__ u16 tile[HW * (CH + 8)];
    const int y = blockIdx.x, n = blockIdx.y;
    const int tid = threadIdx.x;
    // load: [c][x] coalesced over x
    for (int e = tid; e < CH * HW; e += 256) {
        const int c = e >> 6, xx = e & 63;
        const int cn = n * CH + c;
        const float v = x[(size_t)cn * PIX + y * HW + xx];
        const float f = fmaxf(fmaf(v, a[c], b[c]), 0.f) * scale[cn];
        tile[xx * (CH + 8) + c] = f2bf(f);
    }
    __syncthreads();
    // store: [x][c] coalesced over c (u32 = 2 bf16)
    u32* dst = (u32*)(xs2 + ((size_t)n * HW + y) * HW * CH);
    for (int e = tid; e < (CH * HW) / 2; e += 256) {
        const int xx = e >> 8, cp = e & 255;
        const u32 lo = tile[xx * (CH + 8) + 2 * cp];
        const u32 hi = tile[xx * (CH + 8) + 2 * cp + 1];
        dst[xx * (CH / 2) + cp] = lo | (hi << 16);
    }
}

// ---------------- K5: implicit-GEMM 3x3 conv via bf16 WMMA + async LDS staging ----------------
// grid (64 rows, 2 out-ch blocks of 256, 16 samples), block 256 (8 waves)
// wave w: out channels ob+32w..+31 (2 M-tiles), all 64 pixels of row y (4 N-tiles)
// LDS tile (x2 buffers): [row(3)][col(68)][ic(32)] bf16; interior cols staged by
// global_load_async_to_lds_b128 from channel-last xs2; halos/invalid rows pre-zeroed.
__global__ __launch_bounds__(256) void conv_wmma_kernel(
    const u16* __restrict__ xs2,     // [16][64][64][512] bf16 channel-last
    const u16* __restrict__ wpk,     // [9][512][512]     bf16
    const float* __restrict__ demod, // [16][512]
    float* __restrict__ out)         // [16][512][64][64]
{
    __shared__ __align__(32) u16 lds[2][3 * 68 * 32];

    const int y   = blockIdx.x;
    const int ob  = blockIdx.y * 256;
    const int n   = blockIdx.z;
    const int tid = threadIdx.x;
    const int lane  = tid & 31;
    const int wv    = tid >> 5;
    const int lhalf = lane >> 4;      // lane half h holds k 16h..16h+15 (contiguous order)
    const int lm    = lane & 15;

    v8f acc[2][4] = {};               // [M-tile][N-tile], each 16x16 f32

    const u16* xbase = xs2 + (size_t)n * PIX * CH;
    const int  orow0 = ob + wv * 32 + lm;

    // ---- pre-zero cells the async engine never writes (both buffers) ----
    #pragma unroll
    for (int buf = 0; buf < 2; ++buf) {
        for (int u = tid; u < 192; u += 256) {          // halo cols x=-1, x=64
            const int pair = u >> 1, side = u & 1;
            const int icl = pair & 31, r = pair >> 5;
            lds[buf][(r * 68 + (side ? 65 : 0)) * 32 + icl] = 0;
        }
        #pragma unroll
        for (int r = 0; r < 3; ++r) {                   // out-of-image row slabs
            const int gy = y + r - 1;
            if (gy < 0 || gy >= HW)
                for (int j = tid; j < 68 * 32; j += 256) lds[buf][r * 68 * 32 + j] = 0;
        }
    }

    // async stage of one 32-channel chunk into lds[buf]: 3 rows x 64 cols x 64B
    auto stage = [&](int buf, int ic0) {
        const int q = tid & 3, col_i = tid >> 2;        // col_i = x pixel 0..63
        #pragma unroll
        for (int r = 0; r < 3; ++r) {
            const int gy = y + r - 1;                   // block-uniform branch
            if (gy >= 0 && gy < HW) {
                const u32 loff = (u32)(size_t)&lds[buf][(r * 68 + col_i + 1) * 32 + q * 8];
                const u16* gp = xbase + ((size_t)gy * HW + col_i) * CH + ic0 + q * 8;
                async_ld_b128(loff, gp);
            }
        }
    };

    int cur = 0;
    stage(0, 0);
    for (int ic0 = 0; ic0 < CH; ic0 += 32) {
        wait_async0();                // my async transfers for lds[cur] done
        __syncthreads();              // everyone's done; prior compute finished
        if (ic0 + 32 < CH) stage(cur ^ 1, ic0 + 32);

        // ---- 9 taps x (2 M-tiles x 4 N-tiles) of WMMA ----
        #pragma unroll
        for (int dy = 0; dy < 3; ++dy) {
            #pragma unroll
            for (int dx = 0; dx < 3; ++dx) {
                const int tap = dy * 3 + dx;
                const u16* wp0 = wpk + ((size_t)tap * CH + orow0) * CH + ic0 + lhalf * 16;
                const v8bf a0lo = *(const v8bf*)wp0;
                const v8bf a0hi = *(const v8bf*)(wp0 + 8);
                const v8bf a1lo = *(const v8bf*)(wp0 + 16 * CH);
                const v8bf a1hi = *(const v8bf*)(wp0 + 16 * CH + 8);
                const v16bf af0 = __builtin_shufflevector(
                    a0lo, a0hi, 0,1,2,3,4,5,6,7,8,9,10,11,12,13,14,15);
                const v16bf af1 = __builtin_shufflevector(
                    a1lo, a1hi, 0,1,2,3,4,5,6,7,8,9,10,11,12,13,14,15);
                #pragma unroll
                for (int t = 0; t < 4; ++t) {
                    const int col = t * 16 + lm + dx;   // LDS col = pixel + dx
                    const v16bf bf = *(const v16bf*)&lds[cur][(dy * 68 + col) * 32 + lhalf * 16];
                    acc[0][t] = __builtin_amdgcn_wmma_f32_16x16x32_bf16(
                        false, af0, false, bf, (short)0, acc[0][t], false, false);
                    acc[1][t] = __builtin_amdgcn_wmma_f32_16x16x32_bf16(
                        false, af1, false, bf, (short)0, acc[1][t], false, false);
                }
            }
        }
        cur ^= 1;
    }

    // ---- epilogue: apply demod[n,o], store f32 ----
    float* obase = out + (((size_t)n * CH + ob + wv * 32) * HW + y) * HW;
    const float* dmd = demod + n * CH + ob + wv * 32;
    #pragma unroll
    for (int mt = 0; mt < 2; ++mt) {
        #pragma unroll
        for (int r = 0; r < 8; ++r) {
            const int ol = mt * 16 + r + 8 * lhalf;   // C/D layout: VGPR r -> M=r (+8 hi half)
            const float d = dmd[ol];
            float* op = obase + (size_t)ol * PIX + lm;
            op[0]  = acc[mt][0][r] * d;
            op[16] = acc[mt][1][r] * d;
            op[32] = acc[mt][2][r] * d;
            op[48] = acc[mt][3][r] * d;
        }
    }
}

extern "C" void kernel_launch(void* const* d_in, const int* in_sizes, int n_in,
                              void* d_out, int out_size, void* d_ws, size_t ws_size,
                              hipStream_t stream) {
    const float* x      = (const float*)d_in[0];
    const float* y      = (const float*)d_in[1];
    const float* weight = (const float*)d_in[2];
    const float* mod_w  = (const float*)d_in[3];
    const float* mod_b  = (const float*)d_in[4];
    const float* gamma  = (const float*)d_in[5];
    const float* beta   = (const float*)d_in[6];
    float* out = (float*)d_out;

    char* ws = (char*)d_ws;
    float* a_s   = (float*)(ws + 0);          // 512 f32
    float* b_s   = (float*)(ws + 2048);       // 512 f32
    float* scale = (float*)(ws + 4096);       // 16*512 f32
    float* demod = (float*)(ws + 36864);      // 16*512 f32
    float* wsq   = (float*)(ws + 69632);      // 512*512 f32 (1 MB)
    u16*   wpk   = (u16*)  (ws + 1118208);    // 9*512*512 bf16 (4.7 MB)
    u16*   xs2   = (u16*)  (ws + 5836800);    // 16*64*64*512 bf16 (67 MB, channel-last)

    bn_stats_kernel <<<512,  256, 0, stream>>>(x, gamma, beta, a_s, b_s);
    mod_scale_kernel<<<32,   256, 0, stream>>>(y, mod_w, mod_b, scale);
    wsq_kernel      <<<1024, 256, 0, stream>>>(weight, wsq);
    demod_kernel    <<<32,   256, 0, stream>>>(scale, wsq, demod);
    pack_w_kernel   <<<9216, 256, 0, stream>>>(weight, wpk);
    make_xs_kernel  <<<dim3(HW, N_BATCH), 256, 0, stream>>>(x, a_s, b_s, scale, xs2);
    conv_wmma_kernel<<<dim3(HW, 2, N_BATCH), 256, 0, stream>>>(xs2, wpk, demod, out);
}
